// MultiHeadAttention_61323543052320
// MI455X (gfx1250) — matrix operand order repaired and checked by reading, not compiled, not run on hardware
//
#include <hip/hip_runtime.h>
#include <hip/hip_bf16.h>

// MI455X (gfx1250) multi-head attention forward.
// - All FLOPs on v_wmma_f32_16x16x32_bf16 (wave32 WMMA).
// - Flash-style attention (no SxS materialization; avoids ~1 GB HBM traffic
//   that would swamp even 23.3 TB/s).
// - Weight tiles staged into LDS via the Tensor Data Mover (tensor_load_to_lds
//   with D# LDS-padding for bank-friendly B-fragment reads).
// - Fully software-pipelined inner loops: A (global) and B (LDS) fragments are
//   double-buffered so each k-step's 4 WMMAs issue back-to-back with no
//   same-step load waits.

#define DEVFN __device__ __forceinline__

typedef __bf16 bf16_t;
typedef __bf16 v16bf  __attribute__((ext_vector_type(16)));
typedef __bf16 bf16x8 __attribute__((ext_vector_type(8)));
typedef float  v8f    __attribute__((ext_vector_type(8)));
typedef unsigned int u32x4 __attribute__((ext_vector_type(4)));
typedef int          i32x4 __attribute__((ext_vector_type(4)));
typedef int          i32x8 __attribute__((ext_vector_type(8)));

constexpr int BATCH = 2;
constexpr int SEQ   = 2048;
constexpr int DM    = 1024;   // d_model
constexpr int FD    = 64;     // d_feature per head
constexpr int NH    = 16;     // heads
constexpr int BH    = BATCH * NH;
constexpr int BS    = BATCH * SEQ;

// LDS weight tile: 64 rows x 1024 bf16, TDM pads +16B per 1KB of data
// -> row stride 2080 B (1040 elems), rows land on distinct bank groups.
constexpr int    LDS_ROW  = 1040;                       // elems per row incl. pad
constexpr size_t SMEM_W_B = (size_t)FD * LDS_ROW * 2 + 16;

// ---------------- bf16 helpers (round-to-nearest-even) ----------------
DEVFN unsigned short f2bf_bits(float f) {
  unsigned u = __float_as_uint(f);
  u += 0x7fffu + ((u >> 16) & 1u);
  return (unsigned short)(u >> 16);
}
DEVFN bf16_t f2bf(float f) {
  unsigned short h = f2bf_bits(f);
  bf16_t r;
  __builtin_memcpy(&r, &h, 2);
  return r;
}

// ---------------- WMMA fragment loaders (contiguous k = 16*(lane/16) + s in A and B) ----------------
// A-matrix 16x32 bf16, source row-major [M][K], leading dim lda.
DEVFN v16bf load_a_frag(const bf16_t* base, int lda, int row_base, int k_base, int lane) {
  int m  = lane & 15;
  int kh = lane >> 4;
  const bf16_t* p = base + (size_t)(row_base + m) * lda + k_base;
  bf16x8 lo = *(const bf16x8*)(p + 16 * kh);
  bf16x8 hi = *(const bf16x8*)(p + 16 * kh + 8);
  v16bf a;
#pragma unroll
  for (int i = 0; i < 8; ++i) { a[i] = lo[i]; a[8 + i] = hi[i]; }
  return a;
}

// B-matrix 32x16 bf16, source stored [N][K] row-major, leading dim ldb.
DEVFN v16bf load_b_frag(const bf16_t* base, int ldb, int col_base, int k_base, int lane) {
  int n  = lane & 15;
  int kh = lane >> 4;
  const bf16_t* p = base + (size_t)(col_base + n) * ldb + k_base + 16 * kh;
  bf16x8 lo = *(const bf16x8*)(p);
  bf16x8 hi = *(const bf16x8*)(p + 8);
  v16bf b;
#pragma unroll
  for (int i = 0; i < 8; ++i) { b[i] = lo[i]; b[8 + i] = hi[i]; }
  return b;
}

// B-fragment from the TDM-padded LDS tile ([64][1024] data, +8 elems pad per
// 512 data elems -> elem addr = row*1040 + d + (d>=512 ? 8 : 0)).
DEVFN v16bf lds_b_frag(const bf16_t* sW, int col_base, int k_base, int lane) {
  int n  = lane & 15;
  int kh = lane >> 4;
  int d  = k_base + 16 * kh;
  const bf16_t* p = sW + (size_t)(col_base + n) * LDS_ROW + d + (d >= 512 ? 8 : 0);
  bf16x8 lo = *(const bf16x8*)(p);
  bf16x8 hi = *(const bf16x8*)(p + 8);
  v16bf b;
#pragma unroll
  for (int i = 0; i < 8; ++i) { b[i] = lo[i]; b[8 + i] = hi[i]; }
  return b;
}

DEVFN v8f wmma_bf16(v16bf a, v16bf b, v8f c) {
  return __builtin_amdgcn_wmma_f32_16x16x32_bf16(
      false, a, false, b, (short)0, c, false, false);
}

// ---------------- Tensor Data Mover: DMA a [64 x 1024] bf16 slab to LDS ----
// D# per ISA ch.8: group0 = {count/type/lds/global addr}, group1 = dims/tile/
// stride + pad (interval=256 dwords, amount=4 dwords -> +16B per 1KB).
DEVFN void tdm_load_w64x1024(const bf16_t* gsrc, unsigned lds_byte_off) {
  unsigned long long ga = (unsigned long long)(uintptr_t)gsrc;
  u32x4 g0;
  g0[0] = 1u;                                           // count=1 (valid), user mode
  g0[1] = lds_byte_off;                                 // lds_addr
  g0[2] = (unsigned)(ga & 0xffffffffull);               // global_addr[31:0]
  g0[3] = (unsigned)((ga >> 32) & 0x1ffffffull)         // global_addr[56:32]
        | (2u << 30);                                   // type=2 ("image")
  i32x8 g1;
  g1[0] = (1 << 16)        // data_size = 2 bytes
        | (1 << 20)        // pad_enable
        | (7 << 22)        // pad_interval: 256 dwords (1KB)
        | (3 << 25);       // pad_amount: 4 dwords (16B)
  g1[1] = (int)(1024u << 16);                 // tensor_dim0[15:0]  (=1024)
  g1[2] = (int)(64u << 16);                   // dim0 hi=0 | tensor_dim1[15:0]=64
  g1[3] = (int)(1024u << 16);                 // dim1 hi=0 | tile_dim0=1024
  g1[4] = 64;                                 // tile_dim1=64 | tile_dim2=0
  g1[5] = 1024;                               // tensor_dim0_stride[31:0]
  g1[6] = 0;                                  // stride0 hi | stride1 lo
  g1[7] = 0;                                  // stride1 hi
  i32x4 g2 = {0, 0, 0, 0};
  i32x4 g3 = {0, 0, 0, 0};
#if __clang_major__ >= 23
  i32x8 g4 = {0, 0, 0, 0, 0, 0, 0, 0};
  __builtin_amdgcn_tensor_load_to_lds(g0, g1, g2, g3, g4, 0);
#else
  __builtin_amdgcn_tensor_load_to_lds(g0, g1, g2, g3, 0);
#endif
}

// ---------------- prep kernels ----------------
__global__ void cvt_f32_bf16(const float* __restrict__ in, bf16_t* __restrict__ out, int n) {
  for (int i = blockIdx.x * blockDim.x + threadIdx.x; i < n; i += gridDim.x * blockDim.x)
    out[i] = f2bf(in[i]);
}

// [H, D, F] fp32 -> [H, F, D] bf16 so B-fragments read contiguous K.
__global__ void cvt_w_transpose(const float* __restrict__ in, bf16_t* __restrict__ out, int n) {
  for (int i = blockIdx.x * blockDim.x + threadIdx.x; i < n; i += gridDim.x * blockDim.x) {
    int h   = i / (DM * FD);
    int rem = i - h * (DM * FD);
    int d   = rem / FD;
    int f   = rem - d * FD;
    out[((size_t)h * FD + f) * DM + d] = f2bf(in[i]);
  }
}

// ---------------- per-head Q/K/V projection GEMM ----------------
// Out[b,h,s,f] = sum_d X[b,s,d] * W[h,d,f] + bias[h,f]
// grid=(SEQ/64, BH), block=128 (4 waves). W[h] (128KB) staged to LDS by TDM.
// A (global) and B (LDS) fragments double-buffered: the 4 WMMAs of step k
// consume data fetched during step k-1.
__global__ __launch_bounds__(128) void proj_gemm(
    const bf16_t* __restrict__ X,    // [BATCH, SEQ, DM]
    const bf16_t* __restrict__ Wt,   // [NH, FD, DM]
    const float*  __restrict__ bias, // [NH, FD]
    bf16_t*       __restrict__ Out,
    int transpose_out) {
  extern __shared__ bf16_t sW[];
  int bh   = blockIdx.y;
  int b    = bh / NH;
  int h    = bh % NH;
  int wave = threadIdx.x >> 5;
  int lane = threadIdx.x & 31;
  int row  = blockIdx.x * 64 + wave * 16;

  const bf16_t* A = X + (size_t)b * SEQ * DM;

  if (wave == 0) {
    tdm_load_w64x1024(Wt + (size_t)h * FD * DM, (unsigned)(uintptr_t)sW);
    __builtin_amdgcn_s_wait_tensorcnt(0);
  }
  __syncthreads();

  v8f acc[4] = {};
  v16bf a  = load_a_frag(A, DM, row, 0, lane);
  v16bf b0 = lds_b_frag(sW, 0,  0, lane);
  v16bf b1 = lds_b_frag(sW, 16, 0, lane);
  v16bf b2 = lds_b_frag(sW, 32, 0, lane);
  v16bf b3 = lds_b_frag(sW, 48, 0, lane);
  for (int k = 0; k < DM; k += 32) {
    int kn = (k + 32 < DM) ? (k + 32) : k;
    v16bf an = load_a_frag(A, DM, row, kn, lane);
    v16bf c0 = lds_b_frag(sW, 0,  kn, lane);
    v16bf c1 = lds_b_frag(sW, 16, kn, lane);
    v16bf c2 = lds_b_frag(sW, 32, kn, lane);
    v16bf c3 = lds_b_frag(sW, 48, kn, lane);
    acc[0] = wmma_bf16(a, b0, acc[0]);
    acc[1] = wmma_bf16(a, b1, acc[1]);
    acc[2] = wmma_bf16(a, b2, acc[2]);
    acc[3] = wmma_bf16(a, b3, acc[3]);
    a = an; b0 = c0; b1 = c1; b2 = c2; b3 = c3;
  }

  int col16 = lane & 15;
  int half  = lane >> 4;
#pragma unroll
  for (int n = 0; n < 4; ++n) {
    int col  = n * 16 + col16;
    float bv = bias[h * FD + col];
    if (!transpose_out) {
#pragma unroll
      for (int r = 0; r < 8; ++r)
        Out[((size_t)bh * SEQ + row + r + 8 * half) * FD + col] = f2bf(acc[n][r] + bv);
    } else {
      bf16x8 v;
#pragma unroll
      for (int r = 0; r < 8; ++r) v[r] = f2bf(acc[n][r] + bv);
      *(bf16x8*)(Out + ((size_t)bh * FD + col) * SEQ + row + 8 * half) = v;
    }
  }
}

// ---------------- flash attention (computed transposed) ----------------
// Per wave: 16 queries. S^T = K*Q^T (queries in lanes, keys in elems) ->
// softmax = 16-elem per-lane reduce + one shfl_xor(16). O^T = V^T*P^T with
// online rescale. K frags double-buffered; V frags issued early so their
// wait overlaps score-WMMAs + softmax VALU.
__global__ __launch_bounds__(128) void flash_attn(
    const bf16_t* __restrict__ Qp,   // [BH, SEQ, FD]
    const bf16_t* __restrict__ Kp,   // [BH, SEQ, FD]
    const bf16_t* __restrict__ Vt,   // [BH, FD, SEQ]
    bf16_t*       __restrict__ AttnOut) {  // [BATCH, SEQ, DM]
  int bh   = blockIdx.y;
  int b    = bh / NH;
  int h    = bh % NH;
  int wave = threadIdx.x >> 5;
  int lane = threadIdx.x & 31;
  int qb   = blockIdx.x * 64 + wave * 16;

  const bf16_t* Qb = Qp + (size_t)bh * SEQ * FD;
  const bf16_t* Kb = Kp + (size_t)bh * SEQ * FD;
  const bf16_t* Vb = Vt + (size_t)bh * FD * SEQ;

  v16bf qf0 = load_b_frag(Qb, FD, qb, 0, lane);
  v16bf qf1 = load_b_frag(Qb, FD, qb, 32, lane);

  float m = -1e30f, l = 0.0f;
  v8f o[4] = {};
  const float scale   = 0.125f;     // 1/sqrt(64)
  const bool  hi_half = (lane >= 16);

  // prime K fragments for tile 0
  v16bf kA0 = load_a_frag(Kb, FD, 0,  0,  lane);
  v16bf kA1 = load_a_frag(Kb, FD, 0,  32, lane);
  v16bf kB0 = load_a_frag(Kb, FD, 16, 0,  lane);
  v16bf kB1 = load_a_frag(Kb, FD, 16, 32, lane);

  for (int jb = 0; jb < SEQ; jb += 32) {
    // V^T frags for this tile: issue now, consume after softmax
    v16bf va0 = load_a_frag(Vb, SEQ, 0,  jb, lane);
    v16bf va1 = load_a_frag(Vb, SEQ, 16, jb, lane);
    v16bf va2 = load_a_frag(Vb, SEQ, 32, jb, lane);
    v16bf va3 = load_a_frag(Vb, SEQ, 48, jb, lane);

    v8f s0 = {}, s1 = {};
    s0 = wmma_bf16(kA0, qf0, s0);
    s0 = wmma_bf16(kA1, qf1, s0);
    s1 = wmma_bf16(kB0, qf0, s1);
    s1 = wmma_bf16(kB1, qf1, s1);

    // prefetch next tile's K fragments (overlaps softmax below)
    int jn = (jb + 32 < SEQ) ? (jb + 32) : jb;
    kA0 = load_a_frag(Kb, FD, jn,      0,  lane);
    kA1 = load_a_frag(Kb, FD, jn,      32, lane);
    kB0 = load_a_frag(Kb, FD, jn + 16, 0,  lane);
    kB1 = load_a_frag(Kb, FD, jn + 16, 32, lane);

    // online softmax (lane owns one query column)
    float mloc = -1e30f;
#pragma unroll
    for (int r = 0; r < 8; ++r) {
      s0[r] *= scale;
      s1[r] *= scale;
      mloc = fmaxf(mloc, fmaxf(s0[r], s1[r]));
    }
    mloc = fmaxf(mloc, __shfl_xor(mloc, 16, 32));
    float mnew  = fmaxf(m, mloc);
    float alpha = __expf(m - mnew);

    float p0[8], p1[8];
    float rs = 0.0f;
#pragma unroll
    for (int r = 0; r < 8; ++r) {
      p0[r] = __expf(s0[r] - mnew);
      p1[r] = __expf(s1[r] - mnew);
      rs += p0[r] + p1[r];
    }
    rs += __shfl_xor(rs, 16, 32);
    l = l * alpha + rs;
    m = mnew;

    // P^T as 32x16 B-fragment: one cross-half shuffle fixes C->B layout
    v16bf pB;
#pragma unroll
    for (int r = 0; r < 8; ++r) {
      float o0 = __shfl_xor(p0[r], 16, 32);
      float o1 = __shfl_xor(p1[r], 16, 32);
      pB[r]     = f2bf(hi_half ? o1    : p0[r]);
      pB[8 + r] = f2bf(hi_half ? p1[r] : o0);
    }

    // O^T rescale + accumulate V^T * P^T
#pragma unroll
    for (int r = 0; r < 8; ++r) {
      o[0][r] *= alpha; o[1][r] *= alpha; o[2][r] *= alpha; o[3][r] *= alpha;
    }
    o[0] = wmma_bf16(va0, pB, o[0]);
    o[1] = wmma_bf16(va1, pB, o[1]);
    o[2] = wmma_bf16(va2, pB, o[2]);
    o[3] = wmma_bf16(va3, pB, o[3]);
  }

  float inv  = 1.0f / l;
  int   q    = qb + (lane & 15);
  int   half = lane >> 4;
#pragma unroll
  for (int f = 0; f < 4; ++f) {
    bf16x8 v;
#pragma unroll
    for (int r = 0; r < 8; ++r) v[r] = f2bf(o[f][r] * inv);
    *(bf16x8*)(AttnOut + ((size_t)b * SEQ + q) * DM + h * FD + f * 16 + 8 * half) = v;
  }
}

// ---------------- output projection: out = A @ Wo^T + bo (fp32) ----------------
// grid=(DM/64, BS/64), block=128. Wo 64-row slab staged to LDS by TDM.
__global__ __launch_bounds__(128) void out_gemm(
    const bf16_t* __restrict__ A,    // [BS, DM]
    const bf16_t* __restrict__ Wo,   // [DM, DM] row-major [j][i]
    const float*  __restrict__ bo,   // [DM]
    float*        __restrict__ Out) { // [BS, DM] fp32
  extern __shared__ bf16_t sW[];
  int wave = threadIdx.x >> 5;
  int lane = threadIdx.x & 31;
  int row  = blockIdx.y * 64 + wave * 16;
  int cb   = blockIdx.x * 64;

  if (wave == 0) {
    tdm_load_w64x1024(Wo + (size_t)cb * DM, (unsigned)(uintptr_t)sW);
    __builtin_amdgcn_s_wait_tensorcnt(0);
  }
  __syncthreads();

  v8f acc[4] = {};
  v16bf a  = load_a_frag(A, DM, row, 0, lane);
  v16bf b0 = lds_b_frag(sW, 0,  0, lane);
  v16bf b1 = lds_b_frag(sW, 16, 0, lane);
  v16bf b2 = lds_b_frag(sW, 32, 0, lane);
  v16bf b3 = lds_b_frag(sW, 48, 0, lane);
  for (int k = 0; k < DM; k += 32) {
    int kn = (k + 32 < DM) ? (k + 32) : k;
    v16bf an = load_a_frag(A, DM, row, kn, lane);
    v16bf c0 = lds_b_frag(sW, 0,  kn, lane);
    v16bf c1 = lds_b_frag(sW, 16, kn, lane);
    v16bf c2 = lds_b_frag(sW, 32, kn, lane);
    v16bf c3 = lds_b_frag(sW, 48, kn, lane);
    acc[0] = wmma_bf16(a, b0, acc[0]);
    acc[1] = wmma_bf16(a, b1, acc[1]);
    acc[2] = wmma_bf16(a, b2, acc[2]);
    acc[3] = wmma_bf16(a, b3, acc[3]);
    a = an; b0 = c0; b1 = c1; b2 = c2; b3 = c3;
  }

  int col16 = lane & 15;
  int half  = lane >> 4;
#pragma unroll
  for (int n = 0; n < 4; ++n) {
    int col  = cb + n * 16 + col16;
    float bv = bo[col];
#pragma unroll
    for (int r = 0; r < 8; ++r)
      Out[(size_t)(row + r + 8 * half) * DM + col] = acc[n][r] + bv;
  }
}

// ---------------- host launch ----------------
extern "C" void kernel_launch(void* const* d_in, const int* in_sizes, int n_in,
                              void* d_out, int out_size, void* d_ws, size_t ws_size,
                              hipStream_t stream) {
  (void)in_sizes; (void)n_in; (void)out_size; (void)ws_size;
  const float* q_in = (const float*)d_in[0];
  const float* k_in = (const float*)d_in[1];
  const float* v_in = (const float*)d_in[2];
  const float* Wq   = (const float*)d_in[3];
  const float* Wk   = (const float*)d_in[4];
  const float* Wv   = (const float*)d_in[5];
  const float* bq   = (const float*)d_in[6];
  const float* bk   = (const float*)d_in[7];
  const float* bv   = (const float*)d_in[8];
  const float* Wo   = (const float*)d_in[9];
  const float* bo   = (const float*)d_in[10];
  float* out = (float*)d_out;

  char* w = (char*)d_ws;
  size_t off = 0;
  auto carve = [&](size_t elems) {
    bf16_t* p = (bf16_t*)(w + off);
    off += (elems * sizeof(bf16_t) + 255) & ~(size_t)255;
    return p;
  };
  const size_t NX = (size_t)BS * DM;
  const size_t NW = (size_t)NH * DM * FD;
  bf16_t* Xq  = carve(NX);
  bf16_t* Xk  = carve(NX);
  bf16_t* Xv  = carve(NX);
  bf16_t* Wtq = carve(NW);
  bf16_t* Wtk = carve(NW);
  bf16_t* Wtv = carve(NW);
  bf16_t* Wob = carve((size_t)DM * DM);
  bf16_t* Qp  = carve((size_t)BH * SEQ * FD);
  bf16_t* Kp  = carve((size_t)BH * SEQ * FD);
  bf16_t* Vt  = carve((size_t)BH * FD * SEQ);
  bf16_t* AO  = carve(NX);

  dim3 cvb(256), cvg(2048);
  cvt_f32_bf16<<<cvg, cvb, 0, stream>>>(q_in, Xq, (int)NX);
  cvt_f32_bf16<<<cvg, cvb, 0, stream>>>(k_in, Xk, (int)NX);
  cvt_f32_bf16<<<cvg, cvb, 0, stream>>>(v_in, Xv, (int)NX);
  cvt_w_transpose<<<1024, 256, 0, stream>>>(Wq, Wtq, (int)NW);
  cvt_w_transpose<<<1024, 256, 0, stream>>>(Wk, Wtk, (int)NW);
  cvt_w_transpose<<<1024, 256, 0, stream>>>(Wv, Wtv, (int)NW);
  cvt_f32_bf16<<<1024, 256, 0, stream>>>(Wo, Wob, DM * DM);

  dim3 pgrid(SEQ / 64, BH), pblk(128);
  proj_gemm<<<pgrid, pblk, SMEM_W_B, stream>>>(Xq, Wtq, bq, Qp, 0);
  proj_gemm<<<pgrid, pblk, SMEM_W_B, stream>>>(Xk, Wtk, bk, Kp, 0);
  proj_gemm<<<pgrid, pblk, SMEM_W_B, stream>>>(Xv, Wtv, bv, Vt, 1);

  flash_attn<<<dim3(SEQ / 64, BH), dim3(128), 0, stream>>>(Qp, Kp, Vt, AO);

  out_gemm<<<dim3(DM / 64, BS / 64), dim3(128), SMEM_W_B, stream>>>(AO, Wob, bo, out);
}
